// WMSA_V1_41394894799409
// MI455X (gfx1250) — compile-verified, matrix operand order checked by reading, not run on hardware
//
#include <hip/hip_runtime.h>
#include <hip/hip_bf16.h>
#include <stdint.h>
#include <stddef.h>

// ---------------------------------------------------------------------------
// Types / helpers
// ---------------------------------------------------------------------------
typedef __attribute__((ext_vector_type(16))) __bf16 v16bf;
typedef __attribute__((ext_vector_type(8)))  float  v8f;

union AFrag { int4 q[2]; v16bf v; };

__device__ __forceinline__ unsigned short f2bf(float f) {
  union { float f; unsigned u; } c; c.f = f;
  unsigned u = c.u + 0x7FFFu + ((c.u >> 16) & 1u);   // RNE
  return (unsigned short)(u >> 16);
}
__device__ __forceinline__ float bf2f(unsigned short h) {
  union { unsigned u; float f; } c; c.u = ((unsigned)h) << 16;
  return c.f;
}

// Problem constants
#define NROWS   131072        // B * NWIN * P = 2*1024*64
#define ROWS_PB 65536         // rows per batch

// ---------------------------------------------------------------------------
// Tiny utility kernels
// ---------------------------------------------------------------------------
__global__ void zero_f32_k(float* p, int n) {
  int i = blockIdx.x * 256 + threadIdx.x;
  if (i < n) p[i] = 0.f;
}

__global__ void cvt_f32_bf16_k(const float* __restrict__ s, unsigned short* __restrict__ d, int n) {
  int i = blockIdx.x * 256 + threadIdx.x;
  if (i < n) d[i] = f2bf(s[i]);
}

// ---------------------------------------------------------------------------
// Pack x (NCHW fp32, roll -4) -> A rows [row, 256] bf16 (window layout)
// ---------------------------------------------------------------------------
__global__ __launch_bounds__(256) void pack_x_k(const float* __restrict__ x,
                                                unsigned short* __restrict__ A) {
  int t   = blockIdx.x * 256 + threadIdx.x;   // 1,048,576 threads
  int row = t >> 3;
  int c0  = (t & 7) * 32;
  int b   = row >> 16;
  int win = (row >> 6) & 1023;
  int p   = row & 63;
  int yy  = (((win >> 5) << 3) + (p >> 3) + 4) & 255;   // roll(-4): read from +4
  int xx  = (((win & 31) << 3) + (p & 7) + 4) & 255;
  const float* src = x + (((size_t)(b << 8) + c0) << 16) + yy * 256 + xx;
  unsigned short* dst = A + (size_t)row * 256 + c0;
  for (int c = 0; c < 32; ++c)
    dst[c] = f2bf(src[(size_t)c << 16]);
}

// ---------------------------------------------------------------------------
// Generic bf16 WMMA GEMM:  Out[m,n] = sum_k A[m, acol+k] * W[n,k] + bias[n]
//   A: [NROWS, lda] bf16; W: [N, K] bf16 packed; K in {64,128,256}, mult of 32
//   block = 256 thr = 8 waves; wave computes 16 rows x 64 cols (4 WMMA tiles,
//   A fragment reused 4x per K-step; all 4 B tiles loaded as a DS clause
//   before the 4 WMMAs issue). Block tile = 128 rows x 64 cols.
//   N must be a multiple of 64.
//   mode 0: f32 row-major (ldo, ocol); mode 1: bf16 row-major;
//   mode 2: scatter fp32 to NCHW output with roll(+4)
// ---------------------------------------------------------------------------
__global__ __launch_bounds__(256) void gemm_bf16_k(
    const unsigned short* __restrict__ A, int lda, int acol,
    const unsigned short* __restrict__ W, int K,
    const float* __restrict__ bias,
    float* __restrict__ outF, unsigned short* __restrict__ outH,
    int ldo, int ocol, int mode)
{
  __shared__ __align__(16) unsigned short wslab[64 * 256];   // 64 cols x K, 32KB max
  const int tid   = threadIdx.x;
  const int wave  = tid >> 5;
  const int lane  = tid & 31;
  const int nbase = blockIdx.y << 6;
  const int mbase = (blockIdx.x << 7) + (wave << 4);

  // Async-copy the 64 x K weight slab (contiguous since W is [N][K]) to LDS.
  {
    const unsigned short* src = W + (size_t)nbase * K;
    const int chunks = K << 3;                 // (64*K)/8 b128 chunks
    for (int c = tid; c < chunks; c += 256) {
      unsigned ldsa = (unsigned)(size_t)(&wslab[c << 3]);
      const unsigned short* g = src + (c << 3);
      asm volatile("global_load_async_to_lds_b128 %0, %1, off"
                   :: "v"(ldsa), "v"(g) : "memory");
    }
    asm volatile("s_wait_asynccnt 0" ::: "memory");
  }
  __syncthreads();

  const int mlane = lane & 15;
  const int kb0   = (lane < 16) ? 0 : 8;       // per-lane K sub-chunk base
  const int m     = mbase + mlane;
  const unsigned short* arow = A + (size_t)m * lda + acol;
  __builtin_prefetch(arow, 0, 3);

  v8f acc[4];
  #pragma unroll
  for (int nt = 0; nt < 4; ++nt)
    acc[nt] = (v8f){0.f, 0.f, 0.f, 0.f, 0.f, 0.f, 0.f, 0.f};

  for (int k = 0; k < K; k += 32) {
    AFrag a, b[4];
    a.q[0] = *(const int4*)(arow + k + kb0);
    a.q[1] = *(const int4*)(arow + k + kb0 + 16);
    // issue all 8 ds_load_b128 first so LDS latency overlaps the WMMAs
    #pragma unroll
    for (int nt = 0; nt < 4; ++nt) {
      const unsigned short* brow = &wslab[((nt << 4) + mlane) * K + k + kb0];
      b[nt].q[0] = *(const int4*)(brow);
      b[nt].q[1] = *(const int4*)(brow + 16);
    }
    #pragma unroll
    for (int nt = 0; nt < 4; ++nt)
      acc[nt] = __builtin_amdgcn_wmma_f32_16x16x32_bf16(false, a.v, false, b[nt].v,
                                                        (short)0, acc[nt], false, false);
  }

  const int mr0 = mbase + ((lane < 16) ? 0 : 8);
  #pragma unroll
  for (int nt = 0; nt < 4; ++nt) {
    const int   col = nbase + (nt << 4) + mlane;
    const float bv  = bias ? bias[ocol + col] : 0.f;
    if (mode == 2) {                   // scatter to NCHW with roll(+4)
      #pragma unroll
      for (int i = 0; i < 8; ++i) {
        int row = mr0 + i;
        int bb  = row >> 16, win = (row >> 6) & 1023, p = row & 63;
        int yy  = (((win >> 5) << 3) + (p >> 3) + 4) & 255;
        int xx  = (((win & 31) << 3) + (p & 7) + 4) & 255;
        outF[(((size_t)(bb << 8) + (ocol + col)) << 16) + yy * 256 + xx] = acc[nt][i] + bv;
      }
    } else if (mode == 1) {
      #pragma unroll
      for (int i = 0; i < 8; ++i)
        outH[(size_t)(mr0 + i) * ldo + ocol + col] = f2bf(acc[nt][i] + bv);
    } else {
      #pragma unroll
      for (int i = 0; i < 8; ++i)
        outF[(size_t)(mr0 + i) * ldo + ocol + col] = acc[nt][i] + bv;
    }
  }
}

// ---------------------------------------------------------------------------
// GroupNorm stats over q part (qkv cols 0..255), 16 groups of 16 channels
// stats[bg*2] = sum, stats[bg*2+1] = sumsq   (bg = b*16+g)
// ---------------------------------------------------------------------------
__global__ __launch_bounds__(256) void gn_stats_k(const unsigned short* __restrict__ qkv,
                                                  float* __restrict__ stats) {
  __shared__ float ls[256], lss[256];
  int bg = blockIdx.x;                       // 32
  int b = bg >> 4, g = bg & 15;
  size_t rowbase = ((size_t)b << 16) + ((size_t)blockIdx.y << 10);
  float s = 0.f, ss = 0.f;
  for (int i = threadIdx.x; i < 16384; i += 256) {
    size_t r = rowbase + (i >> 4);
    int    c = (g << 4) + (i & 15);
    float  v = bf2f(qkv[r * 768 + c]);
    s += v; ss += v * v;
  }
  ls[threadIdx.x] = s; lss[threadIdx.x] = ss;
  __syncthreads();
  for (int o = 128; o > 0; o >>= 1) {
    if (threadIdx.x < o) { ls[threadIdx.x] += ls[threadIdx.x + o];
                           lss[threadIdx.x] += lss[threadIdx.x + o]; }
    __syncthreads();
  }
  if (threadIdx.x == 0) {
    atomicAdd(&stats[bg * 2 + 0], ls[0]);
    atomicAdd(&stats[bg * 2 + 1], lss[0]);
  }
}

// scal[0] = sum(gn_w); scal[1+bg*2]=mu; scal[2+bg*2]=rstd
__global__ void prep_k(const float* __restrict__ stats, const float* __restrict__ gn_w,
                       float* __restrict__ scal) {
  int t = threadIdx.x;
  if (t == 0) { float s = 0.f; for (int i = 0; i < 256; ++i) s += gn_w[i]; scal[0] = s; }
  if (t < 32) {
    const float N = 16.f * 65536.f;
    float mu  = stats[t * 2] / N;
    float var = stats[t * 2 + 1] / N - mu * mu;
    scal[1 + t * 2] = mu;
    scal[2 + t * 2] = rsqrtf(var + 1e-5f);
  }
}

// ---------------------------------------------------------------------------
// SRU apply (pointwise + channel shuffle) -> q_bf rows [row, 256]
// ---------------------------------------------------------------------------
__global__ __launch_bounds__(256) void sru_k(const unsigned short* __restrict__ qkv,
                                             const float* __restrict__ gn_w,
                                             const float* __restrict__ gn_b,
                                             const float* __restrict__ scal,
                                             unsigned short* __restrict__ qout) {
  int t = blockIdx.x * 256 + threadIdx.x;     // 33,554,432 threads
  int row = t >> 8, c = t & 255;
  int b = row >> 16;
  float gsum = scal[0];
  size_t base = (size_t)row * 768;

  auto gate = [&](int ch, float v, float& w1, float& w2) {
    int bg = (b << 4) + (ch >> 4);
    float mu = scal[1 + bg * 2], rstd = scal[2 + bg * 2];
    float gw = gn_w[ch];
    float gn = (v - mu) * rstd * gw + gn_b[ch];
    float rw = 1.f / (1.f + __expf(-gn * (gw / gsum)));
    w1 = rw > 0.5f ? 1.f : rw;
    w2 = rw > 0.5f ? 0.f : rw;
  };

  int pc = (c < 128) ? c + 128 : c - 128;
  float v  = bf2f(qkv[base + c]);
  float vp = bf2f(qkv[base + pc]);
  float w1c, w2c, w1p, w2p;
  gate(c,  v,  w1c, w2c);
  gate(pc, vp, w1p, w2p);
  qout[(size_t)row * 256 + c] = f2bf(w1c * v + w2p * vp);
}

// ---------------------------------------------------------------------------
// CRU pieces
// ---------------------------------------------------------------------------
// up_sq rows -> spatial NHWC bf16 [b][y][x][64] (rolled frame)
__global__ __launch_bounds__(256) void upsp_k(const unsigned short* __restrict__ upsq,
                                              unsigned short* __restrict__ upsp) {
  int t   = blockIdx.x * 256 + threadIdx.x;   // 131072*4
  int pix = t >> 2, seg = (t & 3) << 4;
  int b = pix >> 16, y = (pix >> 8) & 255, x = pix & 255;
  int row = (b << 16) + ((((y >> 3) << 5) + (x >> 3)) << 6) + ((y & 7) << 3) + (x & 7);
  const int4* s = (const int4*)(upsq + ((size_t)row << 6) + seg);
  int4*       d = (int4*)(upsp + ((size_t)pix << 6) + seg);
  d[0] = s[0]; d[1] = s[1];
}

// 3x3 grouped conv (groups=2, in 64 -> out 256) accumulated into y1 (+= gwc + bias)
__global__ __launch_bounds__(256) void gwc_k(const unsigned short* __restrict__ upsp,
                                             const float* __restrict__ wt,
                                             const float* __restrict__ gbias,
                                             float* __restrict__ y1) {
  int t   = blockIdx.x * 256 + threadIdx.x;   // 131072*32
  int pix = t >> 5;
  int oc0 = (t & 31) << 3;
  int b = pix >> 16, y = (pix >> 8) & 255, x = pix & 255;
  int g = oc0 >> 7;
  float acc[8] = {0.f,0.f,0.f,0.f,0.f,0.f,0.f,0.f};
  for (int ky = 0; ky < 3; ++ky) {
    int yy = y + ky - 1; if (yy < 0 || yy > 255) continue;
    for (int kx = 0; kx < 3; ++kx) {
      int xx = x + kx - 1; if (xx < 0 || xx > 255) continue;
      const unsigned short* src = upsp + (((size_t)(b << 16) + yy * 256 + xx) << 6) + (g << 5);
      const float* wbase = wt + (size_t)oc0 * 32 * 9 + ky * 3 + kx;
      for (int ic = 0; ic < 32; ++ic) {
        float v = bf2f(src[ic]);
        #pragma unroll
        for (int j = 0; j < 8; ++j)
          acc[j] += v * wbase[(size_t)(j * 32 + ic) * 9];
      }
    }
  }
  int row = (b << 16) + ((((y >> 3) << 5) + (x >> 3)) << 6) + ((y & 7) << 3) + (x & 7);
  float* dst = y1 + (size_t)row * 256 + oc0;
  #pragma unroll
  for (int j = 0; j < 8; ++j)
    dst[j] += acc[j] + gbias[oc0 + j];
}

// y2[:,192..255] = low_sq
__global__ void copylow_k(const unsigned short* __restrict__ low, float* __restrict__ y2) {
  int t = blockIdx.x * 256 + threadIdx.x;     // 131072*64
  int row = t >> 6, j = t & 63;
  y2[(size_t)row * 256 + 192 + j] = bf2f(low[((size_t)row << 6) + j]);
}

// channel pooling: pooled[b*512+cc] += partial sums (cc<256 -> y1, else y2)
__global__ __launch_bounds__(256) void pool_k(const float* __restrict__ y1,
                                              const float* __restrict__ y2,
                                              float* __restrict__ pooled) {
  __shared__ float ls[256];
  int bc = blockIdx.x;                        // 1024
  int b = bc >> 9, cc = bc & 511;
  size_t rowbase = ((size_t)b << 16) + ((size_t)blockIdx.y << 11);
  const float* src = (cc < 256) ? (y1 + cc) : (y2 + (cc - 256));
  float s = 0.f;
  for (int i = threadIdx.x; i < 2048; i += 256)
    s += src[(rowbase + i) * 256];
  ls[threadIdx.x] = s;
  __syncthreads();
  for (int o = 128; o > 0; o >>= 1) {
    if (threadIdx.x < o) ls[threadIdx.x] += ls[threadIdx.x + o];
    __syncthreads();
  }
  if (threadIdx.x == 0) atomicAdd(&pooled[bc], ls[0]);
}

__global__ void poolsm_k(const float* __restrict__ pooled, float* __restrict__ sm) {
  __shared__ float red[512];
  int b = blockIdx.x, t = threadIdx.x;        // 2 blocks x 512
  float v = pooled[b * 512 + t] * (1.f / 65536.f);
  red[t] = v; __syncthreads();
  for (int o = 256; o > 0; o >>= 1) { if (t < o) red[t] = fmaxf(red[t], red[t + o]); __syncthreads(); }
  float mx = red[0]; __syncthreads();
  float e = __expf(v - mx);
  red[t] = e; __syncthreads();
  for (int o = 256; o > 0; o >>= 1) { if (t < o) red[t] += red[t + o]; __syncthreads(); }
  sm[b * 512 + t] = e / red[0];
}

// k_final = sm[c]*y1 + sm[256+c]*y2 -> bf16 rows
__global__ __launch_bounds__(256) void cruc_k(const float* __restrict__ y1,
                                              const float* __restrict__ y2,
                                              const float* __restrict__ sm,
                                              unsigned short* __restrict__ kout) {
  int t = blockIdx.x * 256 + threadIdx.x;     // 33,554,432
  int row = t >> 8, c = t & 255;
  int b = row >> 16;
  float v = sm[b * 512 + c] * y1[(size_t)row * 256 + c] +
            sm[b * 512 + 256 + c] * y2[(size_t)row * 256 + c];
  kout[(size_t)row * 256 + c] = f2bf(v);
}

// ---------------------------------------------------------------------------
// Windowed attention: one block per (head, batch, window); 128 thr = 4 waves
// ---------------------------------------------------------------------------
__global__ __launch_bounds__(128) void attn_k(const unsigned short* __restrict__ qb,
                                              const unsigned short* __restrict__ kbuf,
                                              const unsigned short* __restrict__ qkv,
                                              const float* __restrict__ rp,
                                              unsigned short* __restrict__ ob) {
  __shared__ __align__(16) unsigned short qs[64 * 32];
  __shared__ __align__(16) unsigned short ks[64 * 32];
  __shared__ __align__(16) unsigned short vt[32 * 64];   // v transposed [d][r]
  __shared__ float sim[64 * 64];
  __shared__ __align__(16) unsigned short pr[64 * 64];

  int bid = blockIdx.x;
  int h   = bid >> 11;
  int b   = (bid >> 10) & 1;
  int win = bid & 1023;
  size_t rowbase = ((size_t)b << 16) + ((size_t)win << 6);
  int tid = threadIdx.x;

  for (int cchunk = tid; cchunk < 256; cchunk += 128) {
    int r = cchunk >> 2, seg = (cchunk & 3) << 3;
    *(int4*)&qs[(r << 5) + seg] = *(const int4*)(qb   + (rowbase + r) * 256 + (h << 5) + seg);
    *(int4*)&ks[(r << 5) + seg] = *(const int4*)(kbuf + (rowbase + r) * 256 + (h << 5) + seg);
  }
  for (int i = tid; i < 2048; i += 128) {
    int r = i >> 5, d = i & 31;
    vt[(d << 6) + r] = qkv[(rowbase + r) * 768 + 512 + (h << 5) + d];
  }
  __syncthreads();

  int wave = tid >> 5, lane = tid & 31;
  int mlane = lane & 15, kb0 = (lane < 16) ? 0 : 8;

  // phase 1: sim = q @ k^T * scale + rel_bias, masked
  {
    AFrag a;
    int mrow = (wave << 4) + mlane;
    a.q[0] = *(const int4*)&qs[(mrow << 5) + kb0];
    a.q[1] = *(const int4*)&qs[(mrow << 5) + kb0 + 16];
    int wy = win >> 5, wx = win & 31;
    for (int nt = 0; nt < 4; ++nt) {
      AFrag bf;
      int nrow = (nt << 4) + mlane;
      bf.q[0] = *(const int4*)&ks[(nrow << 5) + kb0];
      bf.q[1] = *(const int4*)&ks[(nrow << 5) + kb0 + 16];
      v8f acc = {0.f,0.f,0.f,0.f,0.f,0.f,0.f,0.f};
      acc = __builtin_amdgcn_wmma_f32_16x16x32_bf16(false, a.v, false, bf.v,
                                                    (short)0, acc, false, false);
      int n = (nt << 4) + mlane;
      int jy = n >> 3, jx = n & 7;
      #pragma unroll
      for (int i = 0; i < 8; ++i) {
        int mm = (wave << 4) + i + ((lane < 16) ? 0 : 8);
        int iy = mm >> 3, ix = mm & 7;
        bool masked = ((wy == 31) && ((iy < 4) != (jy < 4))) ||
                      ((wx == 31) && ((ix < 4) != (jx < 4)));
        float bias = rp[h * 225 + (iy - jy + 7) * 15 + (ix - jx + 7)];
        sim[mm * 64 + n] = masked ? -1e30f
                                  : (acc[i] * 0.17677669529663687f + bias);
      }
    }
  }
  __syncthreads();

  // row softmax -> bf16 probs
  if (tid < 64) {
    const float* srow = &sim[tid << 6];
    float mx = -3.4e38f;
    for (int n = 0; n < 64; ++n) mx = fmaxf(mx, srow[n]);
    float s = 0.f;
    for (int n = 0; n < 64; ++n) s += __expf(srow[n] - mx);
    float inv = 1.f / s;
    unsigned short* prow = &pr[tid << 6];
    for (int n = 0; n < 64; ++n) prow[n] = f2bf(__expf(srow[n] - mx) * inv);
  }
  __syncthreads();

  // phase 2: out = probs @ v
  {
    v8f acc0 = {0.f,0.f,0.f,0.f,0.f,0.f,0.f,0.f};
    v8f acc1 = {0.f,0.f,0.f,0.f,0.f,0.f,0.f,0.f};
    int mrow = (wave << 4) + mlane;
    for (int kc = 0; kc < 64; kc += 32) {
      AFrag a, b0, b1;
      a.q[0]  = *(const int4*)&pr[(mrow << 6) + kc + kb0];
      a.q[1]  = *(const int4*)&pr[(mrow << 6) + kc + kb0 + 16];
      b0.q[0] = *(const int4*)&vt[(mlane << 6) + kc + kb0];
      b0.q[1] = *(const int4*)&vt[(mlane << 6) + kc + kb0 + 16];
      b1.q[0] = *(const int4*)&vt[((mlane + 16) << 6) + kc + kb0];
      b1.q[1] = *(const int4*)&vt[((mlane + 16) << 6) + kc + kb0 + 16];
      acc0 = __builtin_amdgcn_wmma_f32_16x16x32_bf16(false, a.v, false, b0.v,
                                                     (short)0, acc0, false, false);
      acc1 = __builtin_amdgcn_wmma_f32_16x16x32_bf16(false, a.v, false, b1.v,
                                                     (short)0, acc1, false, false);
    }
    int mr0 = (wave << 4) + ((lane < 16) ? 0 : 8);
    #pragma unroll
    for (int i = 0; i < 8; ++i) {
      size_t orow = rowbase + mr0 + i;
      ob[orow * 256 + (h << 5) + mlane]      = f2bf(acc0[i]);
      ob[orow * 256 + (h << 5) + 16 + mlane] = f2bf(acc1[i]);
    }
  }
}

// ---------------------------------------------------------------------------
// Host launcher
// ---------------------------------------------------------------------------
extern "C" void kernel_launch(void* const* d_in, const int* in_sizes, int n_in,
                              void* d_out, int out_size, void* d_ws, size_t ws_size,
                              hipStream_t stream) {
  (void)in_sizes; (void)n_in; (void)out_size; (void)ws_size;
  const float* x       = (const float*)d_in[0];
  const float* w_qkv   = (const float*)d_in[1];
  const float* b_qkv   = (const float*)d_in[2];
  const float* rel_pos = (const float*)d_in[3];
  const float* gn_w    = (const float*)d_in[4];
  const float* gn_b    = (const float*)d_in[5];
  const float* sq1_w   = (const float*)d_in[6];
  const float* sq2_w   = (const float*)d_in[7];
  const float* gwc_w   = (const float*)d_in[8];
  const float* gwc_b   = (const float*)d_in[9];
  const float* pwc1_w  = (const float*)d_in[10];
  const float* pwc2_w  = (const float*)d_in[11];
  const float* w_out   = (const float*)d_in[12];
  const float* b_out   = (const float*)d_in[13];

  char* wsb = (char*)d_ws;
  size_t off = 0;
  auto carve = [&](size_t bytes) -> void* {
    void* p = wsb + off;
    off += (bytes + 255) & ~(size_t)255;
    return p;
  };
  unsigned short* Abf   = (unsigned short*)carve((size_t)NROWS * 256 * 2);
  unsigned short* wqkvb = (unsigned short*)carve(768 * 256 * 2);
  unsigned short* woutb = (unsigned short*)carve(256 * 256 * 2);
  unsigned short* sq1b  = (unsigned short*)carve(64 * 128 * 2);
  unsigned short* sq2b  = (unsigned short*)carve(64 * 128 * 2);
  unsigned short* pw1b  = (unsigned short*)carve(256 * 64 * 2);
  unsigned short* pw2b  = (unsigned short*)carve(192 * 64 * 2);
  unsigned short* qkvb  = (unsigned short*)carve((size_t)NROWS * 768 * 2);
  unsigned short* qbf   = (unsigned short*)carve((size_t)NROWS * 256 * 2);
  unsigned short* upsq  = (unsigned short*)carve((size_t)NROWS * 64 * 2);
  unsigned short* lowsq = (unsigned short*)carve((size_t)NROWS * 64 * 2);
  unsigned short* upsp  = (unsigned short*)carve((size_t)NROWS * 64 * 2);
  float*          y1    = (float*)carve((size_t)NROWS * 256 * 4);
  float*          y2    = (float*)carve((size_t)NROWS * 256 * 4);
  unsigned short* kbf   = (unsigned short*)carve((size_t)NROWS * 256 * 2);
  unsigned short* aout  = (unsigned short*)carve((size_t)NROWS * 256 * 2);
  float*          stats = (float*)carve(64 * 4);
  float*          scal  = (float*)carve(128 * 4);
  float*          pooled= (float*)carve(1024 * 4);
  float*          smb   = (float*)carve(1024 * 4);

  // init reductions
  zero_f32_k<<<1, 256, 0, stream>>>(stats, 64);
  zero_f32_k<<<4, 256, 0, stream>>>(pooled, 1024);

  // weight conversion to bf16 [N][K]
  cvt_f32_bf16_k<<<768, 256, 0, stream>>>(w_qkv, wqkvb, 768 * 256);
  cvt_f32_bf16_k<<<256, 256, 0, stream>>>(w_out, woutb, 256 * 256);
  cvt_f32_bf16_k<<<32,  256, 0, stream>>>(sq1_w, sq1b, 64 * 128);
  cvt_f32_bf16_k<<<32,  256, 0, stream>>>(sq2_w, sq2b, 64 * 128);
  cvt_f32_bf16_k<<<64,  256, 0, stream>>>(pwc1_w, pw1b, 256 * 64);
  cvt_f32_bf16_k<<<48,  256, 0, stream>>>(pwc2_w, pw2b, 192 * 64);

  // input pack (roll + window)
  pack_x_k<<<4096, 256, 0, stream>>>(x, Abf);

  // qkv = A @ Wqkv^T + b  -> bf16 [NROWS, 768]   (N=768 -> 12 col-blocks of 64)
  gemm_bf16_k<<<dim3(1024, 12), 256, 0, stream>>>(Abf, 256, 0, wqkvb, 256,
      b_qkv, nullptr, qkvb, 768, 0, 1);

  // SRU on q
  gn_stats_k<<<dim3(32, 64), 256, 0, stream>>>(qkvb, stats);
  prep_k<<<1, 64, 0, stream>>>(stats, gn_w, scal);
  sru_k<<<131072, 256, 0, stream>>>(qkvb, gn_w, gn_b, scal, qbf);

  // CRU on k
  gemm_bf16_k<<<dim3(1024, 1), 256, 0, stream>>>(qkvb, 768, 256, sq1b, 128,
      nullptr, nullptr, upsq, 64, 0, 1);
  gemm_bf16_k<<<dim3(1024, 1), 256, 0, stream>>>(qkvb, 768, 384, sq2b, 128,
      nullptr, nullptr, lowsq, 64, 0, 1);
  upsp_k<<<2048, 256, 0, stream>>>(upsq, upsp);
  gemm_bf16_k<<<dim3(1024, 4), 256, 0, stream>>>(upsq, 64, 0, pw1b, 64,
      nullptr, y1, nullptr, 256, 0, 0);
  gemm_bf16_k<<<dim3(1024, 3), 256, 0, stream>>>(lowsq, 64, 0, pw2b, 64,
      nullptr, y2, nullptr, 256, 0, 0);
  copylow_k<<<32768, 256, 0, stream>>>(lowsq, y2);
  gwc_k<<<16384, 256, 0, stream>>>(upsp, gwc_w, gwc_b, y1);
  pool_k<<<dim3(1024, 32), 256, 0, stream>>>(y1, y2, pooled);
  poolsm_k<<<2, 512, 0, stream>>>(pooled, smb);
  cruc_k<<<131072, 256, 0, stream>>>(y1, y2, smb, kbf);

  // attention
  attn_k<<<16384, 128, 0, stream>>>(qbf, kbf, qkvb, rel_pos, aout);

  // output projection with NCHW + roll(+4) scatter  (N=256 -> 4 col-blocks)
  gemm_bf16_k<<<dim3(1024, 4), 256, 0, stream>>>(aout, 256, 0, woutb, 256,
      b_out, (float*)d_out, nullptr, 0, 0, 2);
}
